// fusion_70841190580372
// MI455X (gfx1250) — matrix operand order repaired and checked by reading, not compiled, not run on hardware
//
#include <hip/hip_runtime.h>

typedef _Float16 v8h  __attribute__((ext_vector_type(8)));
typedef _Float16 v16h __attribute__((ext_vector_type(16)));
typedef float    v8f  __attribute__((ext_vector_type(8)));
typedef unsigned u32x4 __attribute__((ext_vector_type(4)));
typedef unsigned u32x8 __attribute__((ext_vector_type(8)));

#define DGN 8
#define K2N 9

#define GSL(i, n) for (long i = (long)blockIdx.x * blockDim.x + threadIdx.x; i < (n); \
                       i += (long)gridDim.x * blockDim.x)

// LDS byte offset of a __shared__ object (generic -> addrspace(3) -> int)
typedef __attribute__((address_space(3))) void lds_void_t;
static __device__ __forceinline__ unsigned lds_off(const void* p)
{
  return (unsigned)(size_t)(lds_void_t*)p;
}

// ---------------------------------------------------------------------------
// TDM 2D tile load: global (tensorD0 x tensorD1 elements, rowStride elements
// between rows, 4-byte data) -> LDS at ldsByteOff, tile tileD0 x tileD1.
// D# packing per CDNA5 ISA 8.3/8.4 (group0 128b, group1 256b).
// ---------------------------------------------------------------------------
static __device__ __forceinline__ void tdm_load_2d(
    unsigned ldsByteOff, const float* gptr,
    unsigned tensorD0, unsigned tensorD1, unsigned long rowStride,
    unsigned tileD0, unsigned tileD1)
{
  unsigned long ga = (unsigned long)(size_t)gptr;
  u32x4 g0;
  g0[0] = 1u;                                        // count=1, user mode
  g0[1] = ldsByteOff;                                // lds_addr
  g0[2] = (unsigned)(ga & 0xFFFFFFFFu);              // global_addr[31:0]
  g0[3] = (unsigned)((ga >> 32) & 0x1FFFFFFu)        // global_addr[56:32]
        | (2u << 30);                                // type = 2 ("image")
  u32x8 g1;
  g1[0] = 2u << 16;                                  // data_size = 4 bytes
  g1[1] = (tensorD0 & 0xFFFFu) << 16;                // tensor_dim0[15:0] @48
  g1[2] = (tensorD0 >> 16) | ((tensorD1 & 0xFFFFu) << 16);
  g1[3] = (tensorD1 >> 16) | (tileD0 << 16);         // tile_dim0 @112
  g1[4] = tileD1 & 0xFFFFu;                          // tile_dim1; tile_dim2=0
  g1[5] = (unsigned)(rowStride & 0xFFFFFFFFu);       // tensor_dim0_stride lo
  g1[6] = (unsigned)((rowStride >> 32) & 0xFFFFu);   // stride hi; dim1_stride=0
  g1[7] = 0u;
  asm volatile("tensor_load_to_lds %0, %1" :: "s"(g0), "s"(g1) : "memory");
}

// ---------------------------------------------------------------------------
// Generic implicit-GEMM conv, WMMA f32 <= f16 x f16 (wave32 16x16x32).
// Block = 256 threads = 8 waves; output tile = 64 cout x 64 pixels.
// Wave w: cout slice (w>>1)*16, pixel half (w&1)*32 -> 2 WMMAs per K-chunk.
// Cin loop unrolled x2 (all Cin are multiples of 64 on this path).
// ---------------------------------------------------------------------------
__global__ __launch_bounds__(256) void conv_wmma_k(
    const float* __restrict__ in, const float* __restrict__ wgt,
    const float* __restrict__ bias, float* __restrict__ out,
    int Cin, int Cout, int Hin, int Win, int Hout, int Wout,
    int KS, int stride, int pad, int act,
    long inBStride, long inBase, int coutTiles)
{
  const int bz = blockIdx.z;
  const int b  = bz / coutTiles;
  const int coutBase = (bz % coutTiles) * 64;
  const int y  = blockIdx.y;
  const int x0 = blockIdx.x * 64;
  const int tid  = threadIdx.x;
  const int lane = tid & 31;
  const int wave = tid >> 5;
  const int waveM = wave >> 1;   // 0..3 : 16-row cout slice
  const int waveN = wave & 1;    // 0..1 : 32-pixel half

  __shared__ _Float16 Ash[64 * 32];   // [m][k]
  __shared__ _Float16 Bsh[64 * 32];   // [n][k]

  v8f acc0, acc1;
  #pragma unroll
  for (int i = 0; i < 8; ++i) { acc0[i] = 0.0f; acc1[i] = 0.0f; }

  const float* inb = in + inBase + (long)b * inBStride;
  const int Ktap = KS * KS;

  for (int kidx = 0; kidx < Ktap; ++kidx) {
    const int ky = kidx / KS, kx = kidx % KS;
    const int yin = y * stride + ky - pad;
    const bool yok = (yin >= 0) && (yin < Hin);
    for (int ci0 = 0; ci0 < Cin; ci0 += 64) {
      #pragma unroll
      for (int s = 0; s < 2; ++s) {
        const int cib = ci0 + s * 32;
        { // stage A: 64x32 weights. thread t: m=t>>2, k-part=(t&3)*8
          const int m  = tid >> 2;
          const int kp = (tid & 3) * 8;
          int cw = coutBase + m; if (cw >= Cout) cw = Cout - 1;
          const float* wp = wgt + (long)cw * Cin * Ktap + kidx;
          __builtin_prefetch(wp + (long)(cib + 32) * Ktap, 0, 1);
          #pragma unroll
          for (int j = 0; j < 8; ++j)
            Ash[m * 32 + kp + j] = (_Float16)wp[(long)(cib + kp + j) * Ktap];
        }
        { // stage B: 64x32 im2col tile. thread t: n=t>>2, k-part=(t&3)*8
          const int n  = tid >> 2;
          const int kp = (tid & 3) * 8;
          const int xin = (x0 + n) * stride + kx - pad;
          const bool ok = yok && (xin >= 0) && (xin < Win);
          #pragma unroll
          for (int j = 0; j < 8; ++j) {
            float v = ok ? inb[((long)(cib + kp + j) * Hin + yin) * Win + xin] : 0.0f;
            Bsh[n * 32 + kp + j] = (_Float16)v;
          }
        }
        __syncthreads();

        const int khalfA = (lane >= 16) ? 16 : 0; // A: K contiguous per group
        const int kbaseB = (lane >= 16) ? 16 : 0; // B: K contiguous per group
        const int mA  = waveM * 16 + (lane & 15);
        const int nB0 = waveN * 32 + (lane & 15);
        v8h a0 = *(const v8h*)&Ash[mA * 32 + khalfA];
        v8h a1 = *(const v8h*)&Ash[mA * 32 + khalfA + 8];
        v16h Af;
        #pragma unroll
        for (int i = 0; i < 8; ++i) { Af[i] = a0[i]; Af[i + 8] = a1[i]; }

        v8h b0 = *(const v8h*)&Bsh[nB0 * 32 + kbaseB];
        v8h b1 = *(const v8h*)&Bsh[nB0 * 32 + kbaseB + 8];
        v16h Bf;
        #pragma unroll
        for (int i = 0; i < 8; ++i) { Bf[i] = b0[i]; Bf[i + 8] = b1[i]; }
        acc0 = __builtin_amdgcn_wmma_f32_16x16x32_f16(
            false, Af, false, Bf, (short)0, acc0, false, false);

        v8h c0 = *(const v8h*)&Bsh[(nB0 + 16) * 32 + kbaseB];
        v8h c1 = *(const v8h*)&Bsh[(nB0 + 16) * 32 + kbaseB + 8];
        v16h Cf;
        #pragma unroll
        for (int i = 0; i < 8; ++i) { Cf[i] = c0[i]; Cf[i + 8] = c1[i]; }
        acc1 = __builtin_amdgcn_wmma_f32_16x16x32_f16(
            false, Af, false, Cf, (short)0, acc1, false, false);
        __syncthreads();
      }
    }
  }

  // Epilogue: C/D layout — VGPR v: m = v + (lane>=16 ? 8 : 0), n = lane&15.
  const int mofs = (lane >= 16) ? 8 : 0;
  const int nloc = waveN * 32 + (lane & 15);
  const int xo0 = x0 + nloc, xo1 = xo0 + 16;
  #pragma unroll
  for (int v = 0; v < 8; ++v) {
    const int co = coutBase + waveM * 16 + v + mofs;
    if (co < Cout) {
      float r0 = acc0[v] + bias[co];
      float r1 = acc1[v] + bias[co];
      if (act) {
        r0 = (r0 > 0.0f) ? r0 : 0.1f * r0;
        r1 = (r1 > 0.0f) ? r1 : 0.1f * r1;
      }
      const long base = (((long)b * Cout + co) * Hout + y) * Wout;
      if (xo0 < Wout) out[base + xo0] = r0;
      if (xo1 < Wout) out[base + xo1] = r1;
    }
  }
}

// ---------------------------------------------------------------------------
// 1x1 conv / GEMM (Cout=64, stride=1, pad=0): A and B tiles staged by the
// Tensor Data Mover (async DMA -> LDS, TENSORcnt), then f32->f16 conversion
// and the same WMMA fragment path. Tile = 64 cout x 64 pixels.
// ---------------------------------------------------------------------------
__global__ __launch_bounds__(256) void gemm1x1_wmma_k(
    const float* __restrict__ in, const float* __restrict__ wgt,
    const float* __restrict__ bias, float* __restrict__ out,
    int Cin, int H, int W, int act, long inBStride, long inBase)
{
  const int b  = blockIdx.z;
  const int y  = blockIdx.y;
  const int x0 = blockIdx.x * 64;
  const int tid  = threadIdx.x;
  const int lane = tid & 31;
  const int wave = tid >> 5;
  const int waveM = wave >> 1;
  const int waveN = wave & 1;

  __shared__ float    Af32[64 * 32];  // [m][k]   (weights, TDM)
  __shared__ float    Bf32[32 * 64];  // [k][n]   (activations, TDM)
  __shared__ _Float16 Ash[64 * 32];   // [m][k]
  __shared__ _Float16 Bsh[64 * 32];   // [n][k]

  v8f acc0, acc1;
  #pragma unroll
  for (int i = 0; i < 8; ++i) { acc0[i] = 0.0f; acc1[i] = 0.0f; }

  const float* inb = in + inBase + (long)b * inBStride;
  const long HW = (long)H * W;
  const unsigned aOff = lds_off(Af32);
  const unsigned bOff = lds_off(Bf32);

  for (int ci0 = 0; ci0 < Cin; ci0 += 32) {
    if (wave == 0) {
      // weights: 64 rows (cout) x 32 cols, row stride Cin elements
      tdm_load_2d(aOff, wgt + (long)ci0, 32u, 64u, (unsigned long)Cin, 32u, 64u);
      // activations: 32 rows (ci) x 64 px, row stride H*W; zero-fill past W
      tdm_load_2d(bOff, inb + (long)ci0 * HW + (long)y * W + x0,
                  (unsigned)(W - x0), 32u, (unsigned long)HW, 64u, 32u);
      __builtin_amdgcn_s_wait_tensorcnt(0);
    }
    __syncthreads();

    // f32 -> f16 tiles (B transposed to [n][k])
    #pragma unroll
    for (int t = 0; t < 8; ++t) {
      const int idx = tid + t * 256;
      Ash[idx] = (_Float16)Af32[idx];
      const int px = idx >> 5, ci = idx & 31;
      Bsh[idx] = (_Float16)Bf32[ci * 64 + px];
    }
    __syncthreads();

    const int khalfA = (lane >= 16) ? 16 : 0;
    const int kbaseB = (lane >= 16) ? 16 : 0;
    const int mA  = waveM * 16 + (lane & 15);
    const int nB0 = waveN * 32 + (lane & 15);
    v8h a0 = *(const v8h*)&Ash[mA * 32 + khalfA];
    v8h a1 = *(const v8h*)&Ash[mA * 32 + khalfA + 8];
    v16h Af;
    #pragma unroll
    for (int i = 0; i < 8; ++i) { Af[i] = a0[i]; Af[i + 8] = a1[i]; }

    v8h b0 = *(const v8h*)&Bsh[nB0 * 32 + kbaseB];
    v8h b1 = *(const v8h*)&Bsh[nB0 * 32 + kbaseB + 8];
    v16h Bf;
    #pragma unroll
    for (int i = 0; i < 8; ++i) { Bf[i] = b0[i]; Bf[i + 8] = b1[i]; }
    acc0 = __builtin_amdgcn_wmma_f32_16x16x32_f16(
        false, Af, false, Bf, (short)0, acc0, false, false);

    v8h c0 = *(const v8h*)&Bsh[(nB0 + 16) * 32 + kbaseB];
    v8h c1 = *(const v8h*)&Bsh[(nB0 + 16) * 32 + kbaseB + 8];
    v16h Cf;
    #pragma unroll
    for (int i = 0; i < 8; ++i) { Cf[i] = c0[i]; Cf[i + 8] = c1[i]; }
    acc1 = __builtin_amdgcn_wmma_f32_16x16x32_f16(
        false, Af, false, Cf, (short)0, acc1, false, false);
    __syncthreads();
  }

  const int mofs = (lane >= 16) ? 8 : 0;
  const int nloc = waveN * 32 + (lane & 15);
  const int xo0 = x0 + nloc, xo1 = xo0 + 16;
  #pragma unroll
  for (int v = 0; v < 8; ++v) {
    const int co = waveM * 16 + v + mofs;
    float r0 = acc0[v] + bias[co];
    float r1 = acc1[v] + bias[co];
    if (act) {
      r0 = (r0 > 0.0f) ? r0 : 0.1f * r0;
      r1 = (r1 > 0.0f) ? r1 : 0.1f * r1;
    }
    const long base = (((long)b * 64 + co) * H + y) * W;
    if (xo0 < W) out[base + xo0] = r0;
    if (xo1 < W) out[base + xo1] = r1;
  }
}

// ---------------------------------------------------------------------------
// Channel-block copy (concat / strided views)
// ---------------------------------------------------------------------------
__global__ void copy_ch_k(const float* __restrict__ src, float* __restrict__ dst,
                          int Bn, int C, int HW, int dstC, int chOff,
                          long sBStride, long sBase)
{
  long total = (long)Bn * C * HW;
  GSL(i, total) {
    int hw = (int)(i % HW); long r = i / HW;
    int c = (int)(r % C);  int b = (int)(r / C);
    dst[((long)b * dstC + chOff + c) * HW + hw] =
        src[sBase + (long)b * sBStride + (long)c * HW + hw];
  }
}

// ---------------------------------------------------------------------------
// DCN: bilinear gather * sigmoid(mask) -> im2col matrix
// ---------------------------------------------------------------------------
__global__ void dcn_sample_k(const float* __restrict__ x, long xBStride, long xBase,
                             const float* __restrict__ om, float* __restrict__ smp,
                             int Bn, int H, int W)
{
  const long HW = (long)H * W;
  long total = (long)Bn * DGN * K2N * HW;
  GSL(i, total) {
    int hw = (int)(i % HW); long r = i / HW;
    int k = (int)(r % K2N); r /= K2N;
    int g = (int)(r % DGN);
    int b = (int)(r / DGN);
    int yy = hw / W, xx = hw % W;
    const float* omb = om + (long)b * 216 * HW;
    float dy  = omb[(long)(g * 18 + 2 * k) * HW + hw];
    float dx  = omb[(long)(g * 18 + 2 * k + 1) * HW + hw];
    float mv  = omb[(long)(144 + g * 9 + k) * HW + hw];
    float msk = 1.0f / (1.0f + expf(-mv));
    float py = (float)yy + (float)(k / 3 - 1) + dy;
    float px = (float)xx + (float)(k % 3 - 1) + dx;
    float y0f = floorf(py), x0f = floorf(px);
    int y0 = (int)y0f, x0 = (int)x0f;
    float ly = py - y0f, lx = px - x0f;
    float w00 = (1.f - ly) * (1.f - lx), w01 = (1.f - ly) * lx;
    float w10 = ly * (1.f - lx),         w11 = ly * lx;
    float f00 = (y0 >= 0 && y0 < H && x0 >= 0 && x0 < W) ? 1.f : 0.f;
    float f01 = (y0 >= 0 && y0 < H && x0 + 1 >= 0 && x0 + 1 < W) ? 1.f : 0.f;
    float f10 = (y0 + 1 >= 0 && y0 + 1 < H && x0 >= 0 && x0 < W) ? 1.f : 0.f;
    float f11 = (y0 + 1 >= 0 && y0 + 1 < H && x0 + 1 >= 0 && x0 + 1 < W) ? 1.f : 0.f;
    int y0c = min(max(y0, 0), H - 1), y1c = min(max(y0 + 1, 0), H - 1);
    int x0c = min(max(x0, 0), W - 1), x1c = min(max(x0 + 1, 0), W - 1);
    const float* xb = x + xBase + (long)b * xBStride + (long)(g * 8) * HW;
    float* ob = smp + ((long)b * 576 + (long)(g * K2N + k) * 8) * HW + hw;
    #pragma unroll
    for (int c = 0; c < 8; ++c) {
      const float* xc = xb + (long)c * HW;
      float s = w00 * f00 * xc[(long)y0c * W + x0c] + w01 * f01 * xc[(long)y0c * W + x1c]
              + w10 * f10 * xc[(long)y1c * W + x0c] + w11 * f11 * xc[(long)y1c * W + x1c];
      ob[(long)c * HW] = s * msk;
    }
  }
}

// wperm[o][g][k][c] = w[o][g*8+c][k]   (64 x 576)
__global__ void wperm_k(const float* __restrict__ w, float* __restrict__ wp)
{
  int i = blockIdx.x * blockDim.x + threadIdx.x;
  if (i < 64 * 576) {
    int c = i % 8; int r = i / 8;
    int k = r % 9; r /= 9;
    int g = r % 8; int o = r / 8;
    wp[i] = w[(o * 64 + g * 8 + c) * 9 + k];
  }
}

// bilinear x2 upsample (half-pixel centers, edge clamp), optional scale
__global__ void upx2_k(const float* __restrict__ in, float* __restrict__ out,
                       int Bn, int C, int H, int W, float scale)
{
  int Ho = 2 * H, Wo = 2 * W;
  long total = (long)Bn * C * Ho * Wo;
  GSL(i, total) {
    int ox = (int)(i % Wo); long r = i / Wo;
    int oy = (int)(r % Ho); r /= Ho;           // r = b*C + c
    float sy = (oy + 0.5f) * 0.5f - 0.5f;
    float sx = (ox + 0.5f) * 0.5f - 0.5f;
    float y0f = floorf(sy), x0f = floorf(sx);
    float fy = sy - y0f, fx = sx - x0f;
    int y0 = (int)y0f, x0 = (int)x0f;
    int y0c = min(max(y0, 0), H - 1), y1c = min(max(y0 + 1, 0), H - 1);
    int x0c = min(max(x0, 0), W - 1), x1c = min(max(x0 + 1, 0), W - 1);
    const float* p = in + r * (long)H * W;
    float v = (1.f - fy) * ((1.f - fx) * p[(long)y0c * W + x0c] + fx * p[(long)y0c * W + x1c])
            + fy * ((1.f - fx) * p[(long)y1c * W + x0c] + fx * p[(long)y1c * W + x1c]);
    out[i] = v * scale;
  }
}

// 3x3 stride-2 pad-1 pooling; ismax!=0 -> max (pad=-inf), else sum/9 (pad=0)
__global__ void pool3_k(const float* __restrict__ in, float* __restrict__ out,
                        int Bn, int C, int H, int W, int Ho, int Wo, int ismax)
{
  long total = (long)Bn * C * Ho * Wo;
  GSL(i, total) {
    int ox = (int)(i % Wo); long r = i / Wo;
    int oy = (int)(r % Ho); r /= Ho;
    const float* p = in + r * (long)H * W;
    float mx = -3.402823e38f, sm = 0.f;
    for (int dy = 0; dy < 3; ++dy) {
      int yy = oy * 2 - 1 + dy; if (yy < 0 || yy >= H) continue;
      for (int dx = 0; dx < 3; ++dx) {
        int xx = ox * 2 - 1 + dx; if (xx < 0 || xx >= W) continue;
        float v = p[(long)yy * W + xx];
        mx = v > mx ? v : mx; sm += v;
      }
    }
    out[i] = ismax ? mx : sm * (1.0f / 9.0f);
  }
}

// temporal attention: corr = sigmoid(<emb[b,t], emb_ref[b]>_c); af = aligned*corr
__global__ void temporal_k(const float* __restrict__ embref, const float* __restrict__ emb,
                           const float* __restrict__ aligned, float* __restrict__ af,
                           int Bn, int HW)
{
  long total = (long)Bn * 2 * HW;
  GSL(i, total) {
    int hw = (int)(i % HW); long r = i / HW;
    int t = (int)(r & 1); int b = (int)(r >> 1);
    const float* er = embref + (long)b * 64 * HW + hw;
    const float* e  = emb + (long)(b * 2 + t) * 64 * HW + hw;
    float s = 0.f;
    for (int c = 0; c < 64; ++c) s += er[(long)c * HW] * e[(long)c * HW];
    float sg = 1.0f / (1.0f + expf(-s));
    const float* ai = aligned + (long)(b * 2 + t) * 64 * HW + hw;
    float* o = af + ((long)b * 128 + t * 64) * HW + hw;
    for (int c = 0; c < 64; ++c) o[(long)c * HW] = ai[(long)c * HW] * sg;
  }
}

__global__ void add_k(float* __restrict__ dst, const float* __restrict__ src, long n)
{
  GSL(i, n) dst[i] += src[i];
}

__global__ void final_k(const float* __restrict__ feat, const float* __restrict__ attn,
                        const float* __restrict__ aadd, float* __restrict__ out, long n)
{
  GSL(i, n) {
    float s = 1.0f / (1.0f + expf(-attn[i]));
    out[i] = feat[i] * s * 2.0f + aadd[i];
  }
}

// ---------------------------------------------------------------------------
// Host orchestration
// ---------------------------------------------------------------------------
static inline int nb(long n)
{
  long b = (n + 255) / 256;
  if (b < 1) b = 1;
  if (b > 32768) b = 32768;
  return (int)b;
}

enum {
  IN_REF = 0, IN_PRE = 1,
  P_L21_B = 2, P_L21_W, P_L22_B, P_L22_W, P_L31_B, P_L31_W, P_L32_B, P_L32_W,
  P_CASDCN_CONV_B = 10, P_CASDCN_CONV_W, P_CASDCN_OM_B, P_CASDCN_OM_W,
  P_CASOFF1_B = 14, P_CASOFF1_W, P_CASOFF2_B, P_CASOFF2_W,
  P_L1_DCN_CONV_B = 18, P_L1_DCN_CONV_W, P_L1_DCN_OM_B, P_L1_DCN_OM_W,
  P_L1_FEAT_B = 22, P_L1_FEAT_W, P_L1_OFF1_B, P_L1_OFF1_W,
  P_L1_OFF2_B, P_L1_OFF2_W, P_L1_OFF3_B, P_L1_OFF3_W,
  P_L2_DCN_CONV_B = 30, P_L2_DCN_CONV_W, P_L2_DCN_OM_B, P_L2_DCN_OM_W,
  P_L2_FEAT_B, P_L2_FEAT_W, P_L2_OFF1_B, P_L2_OFF1_W,
  P_L2_OFF2_B, P_L2_OFF2_W, P_L2_OFF3_B, P_L2_OFF3_W,
  P_L3_DCN_CONV_B = 42, P_L3_DCN_CONV_W, P_L3_DCN_OM_B, P_L3_DCN_OM_W,
  P_L3_OFF1_B, P_L3_OFF1_W, P_L3_OFF2_B, P_L3_OFF2_W,
  P_FEATFUSION_B = 50, P_FEATFUSION_W, P_SA1_B, P_SA1_W, P_SA2_B, P_SA2_W,
  P_SA3_B, P_SA3_W, P_SA4_B, P_SA4_W, P_SA5_B, P_SA5_W,
  P_SAADD1_B, P_SAADD1_W, P_SAADD2_B, P_SAADD2_W,
  P_SAL1_B = 66, P_SAL1_W, P_SAL2_B, P_SAL2_W, P_SAL3_B, P_SAL3_W,
  P_TA1_B = 72, P_TA1_W, P_TA2_B, P_TA2_W
};

extern "C" void kernel_launch(void* const* d_in, const int* in_sizes, int n_in,
                              void* d_out, int out_size, void* d_ws, size_t ws_size,
                              hipStream_t stream)
{
  (void)in_sizes; (void)n_in; (void)out_size; (void)ws_size;
  const float* refp = (const float*)d_in[IN_REF];
  const float* prep = (const float*)d_in[IN_PRE];
  auto PW = [&](int i) { return (const float*)d_in[i]; };

  float* ws = (float*)d_ws;
  const int B = 2;
  const long HW1 = 128L * 128, HW2 = 64L * 64, HW3 = 32L * 32;

  long arena = 0;
  auto alloc = [&](long n) { long r = arena; arena += n; return r; };
  float* l1      = ws + alloc(4L * 64 * HW1);
  float* l2      = ws + alloc(4L * 64 * HW2);
  float* l3      = ws + alloc(4L * 64 * HW3);
  float* aligned = ws + alloc(2L * 128 * HW1);
  float* s_cat   = ws + alloc(2L * 128 * HW1);
  float* s_off   = ws + alloc(2L * 64 * HW1);
  float* s_off2  = ws + alloc(2L * 64 * HW1);
  float* s_feat  = ws + alloc(2L * 64 * HW1);
  float* s_upo   = ws + alloc(2L * 64 * HW1);
  float* s_upf   = ws + alloc(2L * 64 * HW1);
  float* s_tmp   = ws + alloc(2L * 64 * HW1);
  float* s_om    = ws + alloc(2L * 216 * HW1);
  float* s_smp   = ws + alloc(2L * 576 * HW1);
  float* s_wp    = ws + alloc(64L * 576);

  auto convp = [&](const float* in, const float* w, const float* bias, float* out,
                   int Bn, int Cin, int Cout, int Hin, int Win, int KS, int stride,
                   int pad, int act, long inBStride, long inBase) {
    if (KS == 1 && stride == 1 && pad == 0 && Cout == 64) {
      dim3 g((Win + 63) / 64, Hin, Bn);
      gemm1x1_wmma_k<<<g, 256, 0, stream>>>(in, w, bias, out, Cin, Hin, Win, act,
                                            inBStride, inBase);
      return;
    }
    int Hout = (Hin + 2 * pad - KS) / stride + 1;
    int Wout = (Win + 2 * pad - KS) / stride + 1;
    int ct = (Cout + 63) / 64;
    dim3 g((Wout + 63) / 64, Hout, Bn * ct);
    conv_wmma_k<<<g, 256, 0, stream>>>(in, w, bias, out, Cin, Cout, Hin, Win,
                                       Hout, Wout, KS, stride, pad, act,
                                       inBStride, inBase, ct);
  };
  auto conv = [&](const float* in, int wi, int bi, float* out, int Bn, int Cin,
                  int Cout, int Hin, int Win, int KS, int stride, int pad, int act,
                  long inBStride, long inBase) {
    convp(in, PW(wi), PW(bi), out, Bn, Cin, Cout, Hin, Win, KS, stride, pad, act,
          inBStride, inBase);
  };
  auto copych = [&](const float* src, float* dst, int Bn, int C, long HW, int dstC,
                    int chOff, long sBStride, long sBase) {
    long total = (long)Bn * C * HW;
    copy_ch_k<<<nb(total), 256, 0, stream>>>(src, dst, Bn, C, (int)HW, dstC, chOff,
                                             sBStride, sBase);
  };
  auto pool = [&](const float* in, float* outp, int Bn, int C, int H, int W, int ismax) {
    int Ho = (H - 1) / 2 + 1, Wo = (W - 1) / 2 + 1;
    long total = (long)Bn * C * Ho * Wo;
    pool3_k<<<nb(total), 256, 0, stream>>>(in, outp, Bn, C, H, W, Ho, Wo, ismax);
  };
  auto up2 = [&](const float* in, float* outp, int Bn, int C, int H, int W, float sc) {
    long total = (long)Bn * C * 4L * H * W;
    upx2_k<<<nb(total), 256, 0, stream>>>(in, outp, Bn, C, H, W, sc);
  };

  // --- pyramid: l1=(B*2,64,128,128) batch order [ref0,pre0,ref1,pre1] ---
  copych(refp, l1, B, 64, HW1, 128, 0, 64 * HW1, 0);
  copych(prep, l1, B, 64, HW1, 128, 64, 64 * HW1, 0);
  conv(l1, P_L21_W, P_L21_B, s_tmp, 4, 64, 64, 128, 128, 3, 2, 1, 1, 64 * HW1, 0);
  conv(s_tmp, P_L22_W, P_L22_B, l2, 4, 64, 64, 64, 64, 3, 1, 1, 1, 64 * HW2, 0);
  conv(l2, P_L31_W, P_L31_B, s_tmp, 4, 64, 64, 64, 64, 3, 2, 1, 1, 64 * HW2, 0);
  conv(s_tmp, P_L32_W, P_L32_B, l3, 4, 64, 64, 32, 32, 3, 1, 1, 1, 64 * HW3, 0);

  // --- PCD alignment ---
  auto run_pcd = [&](int nbrT, int alignedChOff) {
    const int OFF1W[4] = {0, P_L1_OFF1_W, P_L2_OFF1_W, P_L3_OFF1_W};
    const int OFF1B[4] = {0, P_L1_OFF1_B, P_L2_OFF1_B, P_L3_OFF1_B};
    const int OFF2W[4] = {0, P_L1_OFF2_W, P_L2_OFF2_W, P_L3_OFF2_W};
    const int OFF2B[4] = {0, P_L1_OFF2_B, P_L2_OFF2_B, P_L3_OFF2_B};
    const int OFF3W[4] = {0, P_L1_OFF3_W, P_L2_OFF3_W, 0};
    const int OFF3B[4] = {0, P_L1_OFF3_B, P_L2_OFF3_B, 0};
    const int OMW[4]   = {0, P_L1_DCN_OM_W, P_L2_DCN_OM_W, P_L3_DCN_OM_W};
    const int OMB[4]   = {0, P_L1_DCN_OM_B, P_L2_DCN_OM_B, P_L3_DCN_OM_B};
    const int DCW[4]   = {0, P_L1_DCN_CONV_W, P_L2_DCN_CONV_W, P_L3_DCN_CONV_W};
    const int DCB[4]   = {0, P_L1_DCN_CONV_B, P_L2_DCN_CONV_B, P_L3_DCN_CONV_B};
    const int FCW[4]   = {0, P_L1_FEAT_W, P_L2_FEAT_W, 0};
    const int FCB[4]   = {0, P_L1_FEAT_B, P_L2_FEAT_B, 0};

    for (int i = 3; i >= 1; --i) {
      int H = 128 >> (i - 1);
      long HW = (long)H * H;
      float* lb = (i == 1) ? l1 : (i == 2) ? l2 : l3;
      long lstr = 2L * 64 * HW;
      long nbrBase = (long)nbrT * 64 * HW;

      copych(lb, s_cat, B, 64, HW, 128, 0, lstr, nbrBase);
      copych(lb, s_cat, B, 64, HW, 128, 64, lstr, 0);
      conv(s_cat, OFF1W[i], OFF1B[i], s_off, B, 128, 64, H, H, 3, 1, 1, 1, 128 * HW, 0);
      float* offFinal;
      if (i == 3) {
        conv(s_off, OFF2W[i], OFF2B[i], s_off2, B, 64, 64, H, H, 3, 1, 1, 1, 64 * HW, 0);
        offFinal = s_off2;
      } else {
        copych(s_off, s_cat, B, 64, HW, 128, 0, 64 * HW, 0);
        copych(s_upo, s_cat, B, 64, HW, 128, 64, 64 * HW, 0);
        conv(s_cat, OFF2W[i], OFF2B[i], s_off2, B, 128, 64, H, H, 3, 1, 1, 1, 128 * HW, 0);
        conv(s_off2, OFF3W[i], OFF3B[i], s_off, B, 64, 64, H, H, 3, 1, 1, 1, 64 * HW, 0);
        offFinal = s_off;
      }
      conv(offFinal, OMW[i], OMB[i], s_om, B, 64, 216, H, H, 3, 1, 1, 0, 64 * HW, 0);
      { long total = (long)B * DGN * K2N * HW;
        dcn_sample_k<<<nb(total), 256, 0, stream>>>(lb, lstr, nbrBase, s_om, s_smp, B, H, H); }
      wperm_k<<<(64 * 576 + 255) / 256, 256, 0, stream>>>(PW(DCW[i]), s_wp);
      int gemmAct = (i == 3) ? 1 : 0;
      float* gemmOut = (i == 3) ? s_feat : s_tmp;
      convp(s_smp, s_wp, PW(DCB[i]), gemmOut, B, 576, 64, H, H, 1, 1, 0, gemmAct,
            576 * HW, 0);
      if (i < 3) {
        copych(gemmOut, s_cat, B, 64, HW, 128, 0, 64 * HW, 0);
        copych(s_upf, s_cat, B, 64, HW, 128, 64, 64 * HW, 0);
        conv(s_cat, FCW[i], FCB[i], s_feat, B, 128, 64, H, H, 3, 1, 1,
             (i == 2) ? 1 : 0, 128 * HW, 0);
      }
      if (i > 1) {
        up2(offFinal, s_upo, B, 64, H, H, 2.0f);
        up2(s_feat, s_upf, B, 64, H, H, 1.0f);
      }
    }
    { int H = 128; long HW = HW1;
      copych(s_feat, s_cat, B, 64, HW, 128, 0, 64 * HW, 0);
      copych(l1, s_cat, B, 64, HW, 128, 64, 2L * 64 * HW, 0);
      conv(s_cat, P_CASOFF1_W, P_CASOFF1_B, s_off, B, 128, 64, H, H, 3, 1, 1, 1, 128 * HW, 0);
      conv(s_off, P_CASOFF2_W, P_CASOFF2_B, s_off2, B, 64, 64, H, H, 3, 1, 1, 1, 64 * HW, 0);
      conv(s_off2, P_CASDCN_OM_W, P_CASDCN_OM_B, s_om, B, 64, 216, H, H, 3, 1, 1, 0, 64 * HW, 0);
      long total = (long)B * DGN * K2N * HW;
      dcn_sample_k<<<nb(total), 256, 0, stream>>>(s_feat, 64 * HW, 0, s_om, s_smp, B, H, H);
      wperm_k<<<(64 * 576 + 255) / 256, 256, 0, stream>>>(PW(P_CASDCN_CONV_W), s_wp);
      convp(s_smp, s_wp, PW(P_CASDCN_CONV_B), s_tmp, B, 576, 64, H, H, 1, 1, 0, 1,
            576 * HW, 0);
      copych(s_tmp, aligned, B, 64, HW, 128, alignedChOff, 64 * HW, 0);
    }
  };

  run_pcd(0, 0);    // a0 = pcd_align(ref_l, ref_l)
  run_pcd(1, 64);   // a1 = pcd_align(nbr_l, ref_l)

  // --- TSA fusion ---
  {
    int H = 128; long HW = HW1;
    float* af     = s_cat;
    float* embref = s_off;
    float* emb    = s_smp;
    float* featT  = s_feat;
    float* attn1  = s_off2;
    float* p1     = s_upo;
    float* p2     = s_upo + 2L * 64 * HW2;
    float* cat2   = s_upf;
    float* attn2  = s_tmp;
    float* al     = s_smp + 4L * 64 * HW1;
    float* pl1    = al + 2L * 64 * HW2;
    float* pl2    = pl1 + 2L * 64 * HW3;
    float* catl   = pl2 + 2L * 64 * HW3;
    float* al2    = catl + 2L * 128 * HW3;
    float* al3    = al2 + 2L * 64 * HW3;
    float* alup   = al3 + 2L * 64 * HW3;
    float* attn3  = alup + 2L * 64 * HW2;
    float* attn4  = attn3 + 2L * 64 * HW2;
    float* attnup = s_om;
    float* attnF  = s_om + 2L * 64 * HW1;
    float* t1b    = s_om + 4L * 64 * HW1;
    float* aadd   = s_off;

    conv(aligned, P_TA1_W, P_TA1_B, embref, B, 64, 64, H, H, 3, 1, 1, 0, 128 * HW, 0);
    conv(aligned, P_TA2_W, P_TA2_B, emb, 4, 64, 64, H, H, 3, 1, 1, 0, 64 * HW, 0);
    temporal_k<<<nb((long)B * 2 * HW), 256, 0, stream>>>(embref, emb, aligned, af, B, (int)HW);

    conv(af, P_FEATFUSION_W, P_FEATFUSION_B, featT, B, 128, 64, H, H, 1, 1, 0, 1, 128 * HW, 0);
    conv(af, P_SA1_W, P_SA1_B, attn1, B, 128, 64, H, H, 1, 1, 0, 1, 128 * HW, 0);
    pool(attn1, p1, B, 64, H, H, 1);
    pool(attn1, p2, B, 64, H, H, 0);
    copych(p1, cat2, B, 64, HW2, 128, 0, 64 * HW2, 0);
    copych(p2, cat2, B, 64, HW2, 128, 64, 64 * HW2, 0);
    conv(cat2, P_SA2_W, P_SA2_B, attn2, B, 128, 64, 64, 64, 1, 1, 0, 1, 128 * HW2, 0);
    conv(attn2, P_SAL1_W, P_SAL1_B, al, B, 64, 64, 64, 64, 1, 1, 0, 1, 64 * HW2, 0);
    pool(al, pl1, B, 64, 64, 64, 1);
    pool(al, pl2, B, 64, 64, 64, 0);
    copych(pl1, catl, B, 64, HW3, 128, 0, 64 * HW3, 0);
    copych(pl2, catl, B, 64, HW3, 128, 64, 64 * HW3, 0);
    conv(catl, P_SAL2_W, P_SAL2_B, al2, B, 128, 64, 32, 32, 3, 1, 1, 1, 128 * HW3, 0);
    conv(al2, P_SAL3_W, P_SAL3_B, al3, B, 64, 64, 32, 32, 3, 1, 1, 1, 64 * HW3, 0);
    up2(al3, alup, B, 64, 32, 32, 1.0f);
    conv(attn2, P_SA3_W, P_SA3_B, attn3, B, 64, 64, 64, 64, 3, 1, 1, 1, 64 * HW2, 0);
    add_k<<<nb(2L * 64 * HW2), 256, 0, stream>>>(attn3, alup, 2L * 64 * HW2);
    conv(attn3, P_SA4_W, P_SA4_B, attn4, B, 64, 64, 64, 64, 1, 1, 0, 1, 64 * HW2, 0);
    up2(attn4, attnup, B, 64, 64, 64, 1.0f);
    conv(attnup, P_SA5_W, P_SA5_B, attnF, B, 64, 64, 128, 128, 3, 1, 1, 0, 64 * HW1, 0);
    conv(attnF, P_SAADD1_W, P_SAADD1_B, t1b, B, 64, 64, 128, 128, 1, 1, 0, 1, 64 * HW1, 0);
    conv(t1b, P_SAADD2_W, P_SAADD2_B, aadd, B, 64, 64, 128, 128, 1, 1, 0, 0, 64 * HW1, 0);
    final_k<<<nb(2L * 64 * HW1), 256, 0, stream>>>(featT, attnF, aadd, (float*)d_out,
                                                   2L * 64 * HW1);
  }
}
